// SelfAttention_58798102282345
// MI455X (gfx1250) — matrix operand order repaired and checked by reading, not compiled, not run on hardware
//
#include <hip/hip_runtime.h>

typedef __attribute__((ext_vector_type(16))) _Float16 v16h;
typedef __attribute__((ext_vector_type(8)))  _Float16 v8h;
typedef __attribute__((ext_vector_type(8)))  float    v8f;
typedef __attribute__((ext_vector_type(4)))  unsigned int v4u;
typedef __attribute__((ext_vector_type(8)))  int v8i;
typedef __attribute__((ext_vector_type(4)))  int v4i;
typedef _Float16 half_t;

#define HIDD 1024
#define NH   8
#define HD   128
#define NB   8
#define LQL  4096
#define LKL  128
#define LDSP 136            // per-wave scratch row stride in halfs
// TDM-staged tile: rows of 1024 halfs (2048B) with 16B pad after each 1024B chunk
#define TROWH 1040          // halfs per staged row (2048B data + 2x16B pad)
#define TILE_BYTES (16 * TROWH * 2)          // 33280 B per staged 16x1024 tile
#define SCR_OFF    (2 * TILE_BYTES)          // scratch starts after the two tiles

// ---------- fragment loaders (16x16x32 f16 WMMA, wave32) ----------
__device__ __forceinline__ v16h load_a_frag(const half_t* __restrict__ src, int ld, int lane) {
  const int m  = lane & 15;
  const int kb = (lane >> 4) << 3;          // 0 or 8
  const half_t* p = src + (size_t)m * ld + kb;
  v16h a;
#pragma unroll
  for (int i = 0; i < 8; ++i) a[i]     = p[i];
#pragma unroll
  for (int i = 0; i < 8; ++i) a[8 + i] = p[16 + i];
  return a;
}

// A-fragment from the TDM-staged, padded LDS tile (pad-aware addressing)
__device__ __forceinline__ v16h load_a_tile(const half_t* __restrict__ sh, int k0, int lane) {
  const int m  = lane & 15;
  const int kb = (lane >> 4) << 3;
  const int k1 = k0 + kb, k2 = k1 + 16;
  const half_t* p1 = sh + m * TROWH + k1 + ((k1 >> 9) << 3);  // +8 halfs pad per 512 halfs
  const half_t* p2 = sh + m * TROWH + k2 + ((k2 >> 9) << 3);
  v16h a;
#pragma unroll
  for (int i = 0; i < 8; ++i) a[i]     = p1[i];
#pragma unroll
  for (int i = 0; i < 8; ++i) a[8 + i] = p2[i];
  return a;
}

__device__ __forceinline__ v16h load_b_frag(const half_t* __restrict__ src, int ld, int lane) {
  const int n  = lane & 15;
  const int kb = (lane >> 4) << 3;          // 0 or 8
  const half_t* p = src + (size_t)n * ld + kb;
  v16h b;
#pragma unroll
  for (int i = 0; i < 8; ++i) b[i]     = p[i];
#pragma unroll
  for (int i = 0; i < 8; ++i) b[8 + i] = p[16 + i];
  return b;
}

__device__ __forceinline__ v8f wmma_f16(v16h a, v16h b, v8f c) {
  return __builtin_amdgcn_wmma_f32_16x16x32_f16(false, a, false, b, (short)0, c, false, false);
}

// ---------- TDM: DMA one 16x1024 f16 tile (row stride 1024) into LDS ----------
// D# group0: count=1 | lds_addr | global_addr | type=2
// D# group1: data_size=2B, pad_enable, pad_interval=256dw(1024B), pad_amount=4dw(16B),
//            tensor_dim0=1024, tensor_dim1=16, tile_dim0=1024, tile_dim1=16, stride0=1024
__device__ __forceinline__ void tdm_load_tile(const half_t* gsrc, unsigned lds_byte_off) {
  const unsigned long long ga = (unsigned long long)gsrc;
  v4u g0 = { 1u, lds_byte_off, (unsigned)(ga & 0xFFFFFFFFull),
             (unsigned)((ga >> 32) & 0x01FFFFFFull) | 0x80000000u };
  v8i g1 = { (int)0x07D10000, 0x04000000, 0x00100000, 0x04000000,
             0x00000010, 0x00000400, 0, 0 };
  v4i z4 = { 0, 0, 0, 0 };
  v8i z8 = { 0, 0, 0, 0, 0, 0, 0, 0 };
  // 6-arg variant (clang-23 / therock-10.0 headers): (g0, g1, g2, g3, ext, cpol)
  __builtin_amdgcn_tensor_load_to_lds(g0, g1, z4, z4, z8, 0);
}

// ---------- fp32 -> f16 bulk convert ----------
__global__ __launch_bounds__(256)
void f32_to_f16(const float* __restrict__ in, half_t* __restrict__ out, long n) {
  const long i = ((long)blockIdx.x * 256 + threadIdx.x) * 8;
  if (i + 8 <= n) {
    const float4 a = *(const float4*)(in + i);
    const float4 b = *(const float4*)(in + i + 4);
    v8h o;
    o[0] = (_Float16)a.x; o[1] = (_Float16)a.y; o[2] = (_Float16)a.z; o[3] = (_Float16)a.w;
    o[4] = (_Float16)b.x; o[5] = (_Float16)b.y; o[6] = (_Float16)b.z; o[7] = (_Float16)b.w;
    *(v8h*)(out + i) = o;
  }
}

// ---------- register-blocked GEMM: Y = X @ W^T + bias, 64x64 per wave ----------
// MODE 0: fp32 out row-major; MODE 1: f16 out row-major; MODE 2: f16 per-head-transposed
template<int MODE>
__global__ __launch_bounds__(32)
void gemm64_xwT(const half_t* __restrict__ X, const half_t* __restrict__ W,
                const float* __restrict__ bias, void* __restrict__ Yv,
                int Kdim, int Nld, int tilesN) {
  const int lane = threadIdx.x & 31;
  const unsigned tile = blockIdx.x;
  const int tm = (int)(tile / (unsigned)tilesN);
  const int tn = (int)(tile % (unsigned)tilesN);
  const size_t m0 = (size_t)tm * 64, n0 = (size_t)tn * 64;

  v8f acc[4][4];
#pragma unroll
  for (int i = 0; i < 4; ++i)
#pragma unroll
    for (int j = 0; j < 4; ++j) acc[i][j] = (v8f){};

  for (int k = 0; k < Kdim; k += 32) {
    v16h a[4], b[4];
#pragma unroll
    for (int i = 0; i < 4; ++i) a[i] = load_a_frag(X + (m0 + 16 * i) * Kdim + k, Kdim, lane);
#pragma unroll
    for (int j = 0; j < 4; ++j) b[j] = load_b_frag(W + (n0 + 16 * j) * Kdim + k, Kdim, lane);
#pragma unroll
    for (int i = 0; i < 4; ++i)
#pragma unroll
      for (int j = 0; j < 4; ++j) acc[i][j] = wmma_f16(a[i], b[j], acc[i][j]);
  }

  const int col = lane & 15, hs = lane >> 4;
#pragma unroll
  for (int j = 0; j < 4; ++j) {
    const size_t n = n0 + 16 * j + col;
    const float bv = bias[n];
#pragma unroll
    for (int i = 0; i < 4; ++i) {
#pragma unroll
      for (int r = 0; r < 8; ++r) {
        const size_t m = m0 + 16 * i + r + 8 * hs;
        if constexpr (MODE == 0) {
          ((float*)Yv)[m * Nld + n] = acc[i][j][r] + bv;
        } else if constexpr (MODE == 1) {
          ((half_t*)Yv)[m * Nld + n] = (_Float16)(acc[i][j][r] + bv);
        } else {
          const int h = (int)n >> 7, d = (int)n & 127;
          const int b = (int)m >> 7, kk = (int)m & 127;
          ((half_t*)Yv)[((size_t)(b * NH + h) * HD + d) * LKL + kk] = (_Float16)(acc[i][j][r] + bv);
        }
      }
    }
  }
}

// ---------- fused attention, 8-wave workgroup = 8 heads sharing one q-tile ----------
// Wave 0 DMAs the 16x1024 query & atten tiles to LDS via the Tensor Data Mover;
// each wave h then runs: proj(Q), proj(atten_out), QK^T/scale + aout, mask,
// softmax, P@V -> Xout (f16).
__global__ __launch_bounds__(256)
void attn_fused(const half_t* __restrict__ query, const half_t* __restrict__ atten,
                const half_t* __restrict__ Wq,   const float* __restrict__ bq,
                const half_t* __restrict__ Watt, const float* __restrict__ batt,
                const half_t* __restrict__ Kp,   const half_t* __restrict__ VpT,
                const int* __restrict__ mask,    half_t* __restrict__ Xout) {
  // single static LDS object (assumed at LDS offset 0 for the TDM lds_addr)
  __shared__ __align__(16) unsigned char smem[SCR_OFF + 8 * 16 * LDSP * 2];
  half_t* shQ = (half_t*)smem;                               // staged query tile
  half_t* shA = (half_t*)(smem + TILE_BYTES);                // staged atten tile

  const int lane = threadIdx.x & 31;
  const int h    = threadIdx.x >> 5;                         // wave id == head
  const unsigned wg = blockIdx.x;                            // B*(LQ/16) groups
  const int qt = (int)(wg & 255);
  const int b  = (int)(wg >> 8);
  const int q0 = qt * 16;
  const int col = lane & 15, hs = lane >> 4;

  half_t* scr = (half_t*)(smem + SCR_OFF) + h * (16 * LDSP); // per-wave Q/P scratch

  if (h == 0) {
    tdm_load_tile(query + ((size_t)b * LQL + q0) * HIDD, 0u);
    tdm_load_tile(atten + ((size_t)b * LQL + q0) * HIDD, (unsigned)TILE_BYTES);
    __builtin_amdgcn_s_wait_tensorcnt((short)0);
  }
  __syncthreads();

  const half_t* wqh = Wq   + (size_t)(h * HD) * HIDD;
  const half_t* wah = Watt + (size_t)(h * HD) * HIDD;

  // ---- projections: qacc = q-tile @ Wq_h^T ; aout = atten-tile @ Watt_h^T ----
  v8f qacc[8], aout[8];
#pragma unroll
  for (int j = 0; j < 8; ++j) { qacc[j] = (v8f){}; aout[j] = (v8f){}; }
  for (int k = 0; k < HIDD; k += 32) {
    const v16h qa = load_a_tile(shQ, k, lane);
    const v16h aa = load_a_tile(shA, k, lane);
#pragma unroll
    for (int j = 0; j < 8; ++j) {
      qacc[j] = wmma_f16(qa, load_b_frag(wqh + (size_t)(j * 16) * HIDD + k, HIDD, lane), qacc[j]);
      aout[j] = wmma_f16(aa, load_b_frag(wah + (size_t)(j * 16) * HIDD + k, HIDD, lane), aout[j]);
    }
  }
  // biases; pre-scale Q by 1/sqrt(HD); park Q (f16) in per-wave scratch
  const float inv_scale = 0.08838834764831845f;  // 1/sqrt(128)
#pragma unroll
  for (int j = 0; j < 8; ++j) {
    const float bqv = bq[h * HD + j * 16 + col];
    const float bav = batt[h * HD + j * 16 + col];
#pragma unroll
    for (int r = 0; r < 8; ++r) {
      aout[j][r] += bav;
      scr[(r + 8 * hs) * LDSP + j * 16 + col] = (_Float16)((qacc[j][r] + bqv) * inv_scale);
    }
  }
  __syncthreads();

  // ---- energy = (Q*invs) @ K^T + atten_out (aout regs used directly as C) ----
  v8f e[8];
#pragma unroll
  for (int j = 0; j < 8; ++j) {
    v8f acc = aout[j];
#pragma unroll
    for (int ks = 0; ks < LKL; ks += 32)
      acc = wmma_f16(load_a_frag(scr + ks, LDSP, lane),
                     load_b_frag(Kp + ((size_t)(b * LKL + j * 16)) * HIDD + h * HD + ks, HIDD, lane),
                     acc);
    e[j] = acc;
  }

  // ---- mask ----
#pragma unroll
  for (int j = 0; j < 8; ++j) {
    if (mask[b * LKL + j * 16 + col] == 0) {
#pragma unroll
      for (int r = 0; r < 8; ++r) e[j][r] = -1.0e10f;
    }
  }

  // ---- softmax over 128 cols: row = (8 regs) x (16 lanes of half-wave) x (8 tiles) ----
#pragma unroll
  for (int r = 0; r < 8; ++r) {
    float lm = -3.0e38f;
#pragma unroll
    for (int j = 0; j < 8; ++j) lm = fmaxf(lm, e[j][r]);
    lm = fmaxf(lm, __shfl_xor(lm, 1, 32));
    lm = fmaxf(lm, __shfl_xor(lm, 2, 32));
    lm = fmaxf(lm, __shfl_xor(lm, 4, 32));
    lm = fmaxf(lm, __shfl_xor(lm, 8, 32));
    float ls = 0.f;
#pragma unroll
    for (int j = 0; j < 8; ++j) { const float p = __expf(e[j][r] - lm); e[j][r] = p; ls += p; }
    ls += __shfl_xor(ls, 1, 32);
    ls += __shfl_xor(ls, 2, 32);
    ls += __shfl_xor(ls, 4, 32);
    ls += __shfl_xor(ls, 8, 32);
    const float rls = 1.0f / ls;
#pragma unroll
    for (int j = 0; j < 8; ++j) e[j][r] *= rls;
  }
  __syncthreads();

  // ---- park P (f16) in scratch, then x = P @ V via VpT (contiguous along kk) ----
#pragma unroll
  for (int j = 0; j < 8; ++j)
#pragma unroll
    for (int r = 0; r < 8; ++r)
      scr[(r + 8 * hs) * LDSP + j * 16 + col] = (_Float16)e[j][r];
  __syncthreads();

#pragma unroll
  for (int j = 0; j < 8; ++j) {
    v8f acc = {};
#pragma unroll
    for (int ks = 0; ks < LKL; ks += 32)
      acc = wmma_f16(load_a_frag(scr + ks, LDSP, lane),
                     load_b_frag(VpT + ((size_t)(b * NH + h) * HD + j * 16) * LKL + ks, LKL, lane),
                     acc);
#pragma unroll
    for (int r = 0; r < 8; ++r)
      Xout[((size_t)b * LQL + q0 + r + 8 * hs) * HIDD + h * HD + j * 16 + col] = (_Float16)acc[r];
  }
}

extern "C" void kernel_launch(void* const* d_in, const int* in_sizes, int n_in,
                              void* d_out, int out_size, void* d_ws, size_t ws_size,
                              hipStream_t stream) {
  const float* query = (const float*)d_in[0];
  const float* key   = (const float*)d_in[1];
  const float* value = (const float*)d_in[2];
  const float* atten = (const float*)d_in[3];
  const int*   mask  = (const int*)d_in[4];
  const float* Wq    = (const float*)d_in[5];
  const float* bq    = (const float*)d_in[6];
  const float* Wk    = (const float*)d_in[7];
  const float* bk    = (const float*)d_in[8];
  const float* Wv    = (const float*)d_in[9];
  const float* bv    = (const float*)d_in[10];
  const float* Watt  = (const float*)d_in[11];
  const float* batt  = (const float*)d_in[12];
  const float* Wfc   = (const float*)d_in[13];
  const float* bfc   = (const float*)d_in[14];
  float* out = (float*)d_out;

  const size_t W_N   = (size_t)HIDD * HIDD;          // 1M
  const size_t KV_N  = (size_t)NB * LKL * HIDD;      // 1M
  const size_t ACT_N = (size_t)NB * LQL * HIDD;      // 33.5M

  half_t* p = (half_t*)d_ws;
  half_t* Wq16   = p; p += W_N;
  half_t* Wk16   = p; p += W_N;
  half_t* Wv16   = p; p += W_N;
  half_t* Watt16 = p; p += W_N;
  half_t* Wfc16  = p; p += W_N;
  half_t* key16  = p; p += KV_N;
  half_t* val16  = p; p += KV_N;
  half_t* qry16  = p; p += ACT_N;
  half_t* att16  = p; p += ACT_N;
  half_t* Kp     = p; p += KV_N;                     // [B,128,1024]
  half_t* VpT    = p; p += KV_N;                     // [B,H,128,128]
  half_t* Xatt   = p; p += ACT_N;                    // [B,4096,1024]

  // ---- one-time fp32 -> f16 conversions ----
  f32_to_f16<<<(unsigned)(W_N   / 2048), 256, 0, stream>>>(Wq,   Wq16,   (long)W_N);
  f32_to_f16<<<(unsigned)(W_N   / 2048), 256, 0, stream>>>(Wk,   Wk16,   (long)W_N);
  f32_to_f16<<<(unsigned)(W_N   / 2048), 256, 0, stream>>>(Wv,   Wv16,   (long)W_N);
  f32_to_f16<<<(unsigned)(W_N   / 2048), 256, 0, stream>>>(Watt, Watt16, (long)W_N);
  f32_to_f16<<<(unsigned)(W_N   / 2048), 256, 0, stream>>>(Wfc,  Wfc16,  (long)W_N);
  f32_to_f16<<<(unsigned)(KV_N  / 2048), 256, 0, stream>>>(key,   key16, (long)KV_N);
  f32_to_f16<<<(unsigned)(KV_N  / 2048), 256, 0, stream>>>(value, val16, (long)KV_N);
  f32_to_f16<<<(unsigned)(ACT_N / 2048), 256, 0, stream>>>(query, qry16, (long)ACT_N);
  f32_to_f16<<<(unsigned)(ACT_N / 2048), 256, 0, stream>>>(atten, att16, (long)ACT_N);

  // ---- K/V projections: M = 1024, N = 1024 -> 16x16 blocks of 64x64 ----
  gemm64_xwT<1><<<16 * 16, 32, 0, stream>>>(key16, Wk16, bk, (void*)Kp,  HIDD, HIDD, 16);
  gemm64_xwT<2><<<16 * 16, 32, 0, stream>>>(val16, Wv16, bv, (void*)VpT, HIDD, HIDD, 16);

  // ---- fused attention: B*(LQ/16) = 2048 workgroups x 8 waves (one per head) ----
  attn_fused<<<NB * (LQL / 16), 256, 0, stream>>>(qry16, att16, Wq16, bq, Watt16, batt,
                                                  Kp, VpT, mask, Xatt);

  // ---- final FC: M = 32768, N = 1024 -> 512x16 blocks of 64x64, fp32 out ----
  gemm64_xwT<0><<<512 * 16, 32, 0, stream>>>(Xatt, Wfc16, bfc, (void*)out, HIDD, HIDD, 16);
}
